// DenseNGCNLayer_48541720379664
// MI455X (gfx1250) — matrix operand order repaired and checked by reading, not compiled, not run on hardware
//
#include <hip/hip_runtime.h>
#include <hip/hip_bf16.h>

#define IN_CH   512
#define OUT_CH  64
#define TILE_M  128
#define KSTEP   32
#define NSTEPS  (IN_CH / KSTEP)

typedef __attribute__((ext_vector_type(16))) __bf16 bf16x16;
typedef __attribute__((ext_vector_type(8)))  float  floatx8;

union FragU {
    unsigned u[8];
    bf16x16  v;
};

// fp32 -> bf16 (round to nearest even), two packed into one dword
__device__ __forceinline__ unsigned short f2bf(float f) {
    unsigned u = __float_as_uint(f);
    u += 0x7FFFu + ((u >> 16) & 1u);
    return (unsigned short)(u >> 16);
}
__device__ __forceinline__ unsigned packbf(float a, float b) {
    return (unsigned)f2bf(a) | ((unsigned)f2bf(b) << 16);
}

// ---------------------------------------------------------------------------
// Pack W [512,64] fp32 row-major -> bf16 k-pair-major: wp[(k/2)*64 + n] holds
// {bf16(W[k][n]), bf16(W[k+1][n])} so a B-fragment pair is one dword load.
// ---------------------------------------------------------------------------
__global__ void packW_kernel(const float* __restrict__ w, unsigned* __restrict__ wp) {
    int p = blockIdx.x * blockDim.x + threadIdx.x;   // 0 .. 256*64-1
    if (p >= (IN_CH / 2) * OUT_CH) return;
    int k2 = p / OUT_CH;
    int n  = p - k2 * OUT_CH;
    float a = w[(2 * k2)     * OUT_CH + n];
    float b = w[(2 * k2 + 1) * OUT_CH + n];
    wp[p] = packbf(a, b);
}

// ---------------------------------------------------------------------------
// Y[nRows,64] = X[nRows,512] @ W  via v_wmma_f32_16x16x32_bf16.
// 512 threads = 16 waves; wave (msub,nsub) computes TWO 16x16 tiles
// (msub and msub+4) of the block's 128x64 output tile, reusing its B frag.
// A tile is double-buffered in LDS (one barrier per K-step).
// Out-of-range rows are CLAMPED on load (their C rows are never stored),
// keeping staging loads unconditional and coalesced.
// ---------------------------------------------------------------------------
__global__ __launch_bounds__(512)
void gemm_bf16_wmma_kernel(const float* __restrict__ X,
                           const unsigned* __restrict__ Wp,
                           float* __restrict__ Y, int nRows) {
    __shared__ unsigned aTile[2][TILE_M * 16];   // 2 x (128 rows x 16 pairs) = 16 KB

    const int tid   = threadIdx.x;
    const int wave  = tid >> 5;
    const int lane  = tid & 31;
    const int msub  = wave & 3;      // 0..3 : first M sub-tile (second is +4)
    const int nsub  = wave >> 2;     // 0..3 : N sub-tile
    const int lhalf = lane >> 4;     // 0 | 1 (K-half selector per ISA layout)
    const int lmod  = lane & 15;
    const int r0    = blockIdx.x * TILE_M;
    const int rmax  = nRows - 1;

    auto stage = [&](int k0, int which) {
#pragma unroll
        for (int it = 0; it < (TILE_M * 16) / 512; ++it) {
            int p   = tid + it * 512;
            int row = p >> 4;
            int kp  = p & 15;
            int gr  = min(r0 + row, rmax);            // clamp: no divergence
            float2 f2 = *(const float2*)(X + (size_t)gr * IN_CH + k0 + 2 * kp);
            aTile[which][p] = packbf(f2.x, f2.y);
        }
    };

    floatx8 acc0 = {};
    floatx8 acc1 = {};

    stage(0, 0);
    __syncthreads();

    for (int ks = 0; ks < NSTEPS; ++ks) {
        const int k0  = ks * KSTEP;
        const int buf = ks & 1;

        // --- A fragments (16-bit A 16x32 layout):
        // element i (=2j,2j+1): k = (i>=8?16:0) + (lane>=16?8:0) + i%8
        FragU a0, a1, b;
#pragma unroll
        for (int j = 0; j < 8; ++j) {
            int pairsel = (j & 3) + (lhalf << 2) + ((j >= 4) ? 8 : 0);
            a0.u[j] = aTile[buf][(msub * 16 + lmod) * 16 + pairsel];
            a1.u[j] = aTile[buf][((msub + 4) * 16 + lmod) * 16 + pairsel];
        }
        // --- B fragment (16-bit B 32x16 layout):
        // element i: k = (i>=8?16:0) + (lane>=16?8:0) + i%8, n = lane%16
#pragma unroll
        for (int j = 0; j < 8; ++j) {
            int kpair = (k0 >> 1) + (j & 3) + (lhalf << 2) + ((j >= 4) ? 8 : 0);
            b.u[j] = Wp[kpair * OUT_CH + nsub * 16 + lmod];
        }

        acc0 = __builtin_amdgcn_wmma_f32_16x16x32_bf16(
                   false, a0.v, false, b.v, (short)0, acc0, false, false);
        acc1 = __builtin_amdgcn_wmma_f32_16x16x32_bf16(
                   false, a1.v, false, b.v, (short)0, acc1, false, false);

        if (ks + 1 < NSTEPS)
            stage((ks + 1) * KSTEP, buf ^ 1);
        __syncthreads();
    }

    // --- store C (32-bit C/D 16x16 layout: vgpr j -> M = j + 8*lhalf, N = lmod)
    const int n      = nsub * 16 + lmod;
    const int mbase0 = msub * 16 + lhalf * 8;
    const int mbase1 = (msub + 4) * 16 + lhalf * 8;
#pragma unroll
    for (int j = 0; j < 8; ++j) {
        int gr = r0 + mbase0 + j;
        if (gr < nRows) Y[(size_t)gr * OUT_CH + n] = acc0[j];
    }
#pragma unroll
    for (int j = 0; j < 8; ++j) {
        int gr = r0 + mbase1 + j;
        if (gr < nRows) Y[(size_t)gr * OUT_CH + n] = acc1[j];
    }
}

// ---------------------------------------------------------------------------
// SPMM: out[row[e]] += val[e] * x[col[e]]   (one wave32 per edge, float2/lane)
// x and out are L2-resident (12.8 MB each); atomics resolve in L2.
// ---------------------------------------------------------------------------
__global__ __launch_bounds__(256)
void spmm_kernel(const int* __restrict__ rows, const int* __restrict__ cols,
                 const float* __restrict__ vals, const float* __restrict__ x,
                 float* __restrict__ out, int nEdges) {
    int t    = blockIdx.x * blockDim.x + threadIdx.x;
    int e    = t >> 5;
    int lane = t & 31;
    if (e >= nEdges) return;
    int   r = rows[e];
    int   c = cols[e];
    float v = vals[e];
    float2 g = *(const float2*)(x + (size_t)c * OUT_CH + 2 * lane);
    float* dst = out + (size_t)r * OUT_CH + 2 * lane;
    atomicAdd(dst,     v * g.x);
    atomicAdd(dst + 1, v * g.y);
}

// ---------------------------------------------------------------------------
__global__ void zero_kernel(float4* __restrict__ p, int nVec4) {
    int i = blockIdx.x * blockDim.x + threadIdx.x;
    if (i < nVec4) p[i] = make_float4(0.f, 0.f, 0.f, 0.f);
}

__global__ void bias_kernel(const float* __restrict__ src,
                            const float* __restrict__ bias,
                            float* __restrict__ dst, int n) {
    int i = blockIdx.x * blockDim.x + threadIdx.x;
    if (i < n) dst[i] = src[i] + bias[i & (OUT_CH - 1)];
}

// ---------------------------------------------------------------------------
extern "C" void kernel_launch(void* const* d_in, const int* in_sizes, int n_in,
                              void* d_out, int out_size, void* d_ws, size_t ws_size,
                              hipStream_t stream) {
    const int*   adj   = (const int*)d_in[0];      // [2, E]
    const float* vals  = (const float*)d_in[1];    // [E]
    const float* feats = (const float*)d_in[2];    // [N, 512]
    const float* wmat  = (const float*)d_in[3];    // [512, 64]
    const float* bias  = (const float*)d_in[4];    // [1, 64]

    const int nEdges = in_sizes[1];
    const int nNodes = in_sizes[2] / IN_CH;
    const int nOut   = nNodes * OUT_CH;            // == out_size

    const int* rows = adj;
    const int* cols = adj + nEdges;

    float*    bufA  = (float*)d_ws;                               // N*64 fp32
    unsigned* wpack = (unsigned*)((char*)d_ws + (size_t)nOut * sizeof(float));
    float*    outF  = (float*)d_out;

    // 1) pack W to bf16 pair-major
    {
        int n = (IN_CH / 2) * OUT_CH;
        packW_kernel<<<(n + 255) / 256, 256, 0, stream>>>(wmat, wpack);
    }
    // 2) base = X @ W  (WMMA bf16)  -> bufA
    {
        int grid = (nNodes + TILE_M - 1) / TILE_M;
        gemm_bf16_wmma_kernel<<<grid, 512, 0, stream>>>(feats, wpack, bufA, nNodes);
    }
    // 3) spmm iteration 1: d_out = A @ bufA
    {
        int nv4 = nOut / 4;
        zero_kernel<<<(nv4 + 255) / 256, 256, 0, stream>>>((float4*)outF, nv4);
        long long th = (long long)nEdges * 32;
        spmm_kernel<<<(int)((th + 255) / 256), 256, 0, stream>>>(rows, cols, vals,
                                                                 bufA, outF, nEdges);
    }
    // 4) spmm iteration 2: bufA = A @ d_out
    {
        int nv4 = nOut / 4;
        zero_kernel<<<(nv4 + 255) / 256, 256, 0, stream>>>((float4*)bufA, nv4);
        long long th = (long long)nEdges * 32;
        spmm_kernel<<<(int)((th + 255) / 256), 256, 0, stream>>>(rows, cols, vals,
                                                                 outF, bufA, nEdges);
    }
    // 5) d_out = bufA + bias
    bias_kernel<<<(nOut + 255) / 256, 256, 0, stream>>>(bufA, bias, outF, nOut);
}
